// CorrelationModule_87385404604740
// MI455X (gfx1250) — matrix operand order repaired and checked by reading, not compiled, not run on hardware
//
#include <hip/hip_runtime.h>
#include <stdint.h>

typedef _Float16 v16h __attribute__((ext_vector_type(16)));
typedef _Float16 v8h  __attribute__((ext_vector_type(8)));
typedef _Float16 h2   __attribute__((ext_vector_type(2)));
typedef float    v8f  __attribute__((ext_vector_type(8)));

#define CSTR 136      // halves per LDS column (128 channels + 8 pad vs bank conflicts)
#define NCOL 160      // 148 padded x1p columns + 12 extra zero cols for last M-tile
#define CH   128
#define WH   128
#define DWIN 21

__global__ __launch_bounds__(256)
void corr_wmma_kernel(const float* __restrict__ x1,
                      const float* __restrict__ x2,
                      float* __restrict__ out)
{
    __shared__ __align__(16) _Float16 x1s[NCOL * CSTR];   // 43,520 B: x1p row, [col][c] f16
    __shared__ __align__(16) float outs[(DWIN + 1) * WH]; // 11,264 B: band + dump row 21

    const int tid  = threadIdx.x;
    const int lane = tid & 31;
    const int wv   = tid >> 5;       // wave 0..7 -> N-tile
    const int hf   = lane >> 4;      // lane half (0/1)
    const int n    = lane & 15;      // N index within tile

    const int bi = blockIdx.x;       // 512 blocks = (b, i)
    const int b  = bi >> 7;
    const int i  = bi & 127;
    const int j0 = wv << 4;

    // ---- preload B operand (x2[b,:,i,:]) into registers, WMMA B layout ----
    // B[k,n] for k = 32*kk + 16*hf + h, column j0+n
    v16h breg[4];
    {
        const float* x2b = x2 + (((long)b * CH) * WH + i) * WH; // + c*16384 + j
        const int j = j0 + n;
        #pragma unroll
        for (int kk = 0; kk < 4; ++kk) {
            #pragma unroll
            for (int h = 0; h < 16; ++h) {
                int c = kk * 32 + hf * 16 + h;
                breg[kk][h] = (_Float16)x2b[(long)c * (WH * WH) + j];
            }
        }
    }

    // ---- zero the permanently-zero LDS columns: 0..9 and 138..159, c in [0,128) ----
    {
        v8h z = {};
        #pragma unroll
        for (int k = 0; k < 2; ++k) {
            int idx   = tid + (k << 8);            // 0..511
            int colid = idx >> 4;                  // 0..31
            int seg   = idx & 15;                  // 16B segment
            int col   = (colid < 10) ? colid : colid + 128;  // 0..9, 138..159
            *(v8h*)&x1s[col * CSTR + seg * 8] = z;
        }
    }

    const long x1base = ((long)b * CH) * WH * WH;

    for (int di = 0; di < DWIN; ++di) {
        const int  row   = i + di - 10;            // x1 W-row (x1p row i+di)
        const bool valid = (row >= 0) && (row < WH);

        // ---- stage x1 row -> x1s transposed [col = j+10][c], f16, zero if OOB ----
        // (prev iteration's x1s readers are behind the post-band barrier)
        #pragma unroll
        for (int it = 0; it < 8; ++it) {
            int idx = tid + (it << 8);
            int jg  = idx & 31;                    // lane-contiguous j groups
            int cp  = idx >> 5;                    // channel pair 0..63
            int c   = cp * 2;
            int j   = jg * 4;
            float4 p0 = make_float4(0.f, 0.f, 0.f, 0.f);
            float4 p1 = p0;
            if (valid) {
                const float* r0 = x1 + x1base + ((long)c * WH + row) * WH + j;
                p0 = *(const float4*)r0;
                p1 = *(const float4*)(r0 + WH * WH);
            }
            float a0[4] = {p0.x, p0.y, p0.z, p0.w};
            float a1[4] = {p1.x, p1.y, p1.z, p1.w};
            #pragma unroll
            for (int s = 0; s < 4; ++s) {
                h2 t2;
                t2[0] = (_Float16)a0[s];
                t2[1] = (_Float16)a1[s];
                *(h2*)&x1s[(j + s + 10) * CSTR + c] = t2;   // ds_store_b32
            }
        }
        __syncthreads();                            // B1: stage -> compute

        // ---- band GEMM: 3 M-tiles x (K=128 as 4 steps of 32), accs held in regs ----
        v8f acc[3];
        #pragma unroll
        for (int t = 0; t < 3; ++t) {
            const _Float16* abase = &x1s[(j0 + t * 16 + n) * CSTR + hf * 16];
            v8h lo[4], hi[4];
            #pragma unroll
            for (int kk = 0; kk < 4; ++kk) {        // group loads so they pipeline
                lo[kk] = *(const v8h*)(abase + kk * 32);       // K = 32kk+16hf .. +7
                hi[kk] = *(const v8h*)(abase + kk * 32 + 8);   // K = +8
            }
            v8f a8 = {};
            acc[t] = a8;
            #pragma unroll
            for (int kk = 0; kk < 4; ++kk) {
                v16h a = __builtin_shufflevector(lo[kk], hi[kk],
                          0, 1, 2, 3, 4, 5, 6, 7, 8, 9, 10, 11, 12, 13, 14, 15);
                acc[t] = __builtin_amdgcn_wmma_f32_16x16x32_f16(
                             false, a, false, breg[kk], (short)0, acc[t], false, false);
            }
        }

        // ---- drain last iteration's async writeback before touching outs ----
        asm volatile("s_wait_asynccnt 0x0" ::: "memory");
        __syncthreads();                            // Bb: workgroup-wide async drain

        // ---- band extraction: unconditional stores, invalid dj -> dump row 21 ----
        // D[r,n] = G[j0+16t+r, j0+n], dj = 16t + r - n, r = v + 8*hf
        #pragma unroll
        for (int t = 0; t < 3; ++t) {
            #pragma unroll
            for (int v = 0; v < 8; ++v) {
                int dj = t * 16 + v + hf * 8 - n;
                int r  = ((unsigned)dj <= 20u) ? dj : DWIN;   // dump row
                outs[r * WH + j0 + n] = acc[t][v];
            }
        }
        __syncthreads();                            // Bc: band visible -> async + restage

        // ---- async LDS -> global writeback (overlaps next stage + compute) ----
        const long obase = (((long)(b * (DWIN * DWIN) + di * DWIN)) * WH + i) * WH;
        for (int e = tid; e < DWIN * (WH / 4); e += 256) {    // 672 x b128 chunks
            int dj  = e >> 5;                       // 32 chunks per dj row
            int seg = e & 31;
            const float* gp = out + obase + (long)dj * (WH * WH) + seg * 4;
            uint32_t lp = (uint32_t)(uintptr_t)&outs[dj * WH + seg * 4];
            asm volatile("global_store_async_from_lds_b128 %0, %1, off"
                         :: "v"(gp), "v"(lp) : "memory");
        }
    }
    asm volatile("s_wait_asynccnt 0x0" ::: "memory");
}

extern "C" void kernel_launch(void* const* d_in, const int* in_sizes, int n_in,
                              void* d_out, int out_size, void* d_ws, size_t ws_size,
                              hipStream_t stream) {
    const float* x1 = (const float*)d_in[0];
    const float* x2 = (const float*)d_in[1];
    float* out = (float*)d_out;
    (void)in_sizes; (void)n_in; (void)out_size; (void)d_ws; (void)ws_size;
    corr_wmma_kernel<<<dim3(4 * 128), dim3(256), 0, stream>>>(x1, x2, out);
}
